// QueryEncoder_57277683859810
// MI455X (gfx1250) — matrix operand order repaired and checked by reading, not compiled, not run on hardware
//
#include <hip/hip_runtime.h>

// Problem constants (match reference)
#define BS   64
#define LCT  512
#define NPH  16
#define NQ   16
#define HD   1024
#define EPS  1e-5f

typedef __attribute__((ext_vector_type(16))) _Float16 v16h;
typedef __attribute__((ext_vector_type(2)))  _Float16 v2h;
typedef __attribute__((ext_vector_type(8)))  float    v8f;

static __device__ __forceinline__ float wave_sum(float v) {
#pragma unroll
  for (int off = 16; off > 0; off >>= 1) v += __shfl_xor(v, off, 32);
  return v;
}

// ---------------------------------------------------------------------------
// WMMA GEMM: C[M,N] = A[M,K] @ B[K,N] (+bias).  fp32 in/out, f16 compute,
// f32 accumulate via V_WMMA_F32_16X16X32_F16.
//
// Workgroup tile: (16*nwaves) x 64.  Per K-step of 32, all threads stage the
// 32x64 B panel into LDS *pre-swizzled into WMMA B-fragment order*:
//   B[k][n] -> Bs[(n>>4)*512 + (k>>4)*256 + (n&15)*16 + (k&15)]   (halves)
// so each lane's v16h fragment for subtile s is 32 contiguous bytes:
//   &Bs[s*512 + lane*16]        (ds_load_b128 x2, no gathers)
// The panel is shared by all waves in the WG.  A fragments are read from
// global as coalesced float2 (rows are contiguous in K).
// BT=true reads B as B[k][n] = Bmem[n*ldB + k] (B = W^T, contiguous in k).
// grid = (N/64, ceil(M/(16*nwaves)), batch), block = 32*nwaves.
// ---------------------------------------------------------------------------
template <bool BT>
__global__ void gemm_wmma_f16(const float* __restrict__ A,
                              const float* __restrict__ B,
                              const float* __restrict__ bias,
                              float* __restrict__ C,
                              int M, int N, int K, long ldA, long ldB,
                              long strideA, long strideB, long strideC)
{
  __shared__ __align__(32) _Float16 Bs[32 * 64];   // 4 KB staged B panel

  const int tid  = threadIdx.x;
  const int lane = tid & 31;
  const int w    = tid >> 5;
  const int nw   = blockDim.x >> 5;
  const int mr   = lane & 15;
  const int g    = lane >> 4;

  const long zb = blockIdx.z;
  const float* Ab = A + zb * strideA;
  const float* Bb = B + zb * strideB;
  float*       Cb = C + zb * strideC;

  const int m0 = blockIdx.y * (nw << 4) + (w << 4);
  const int n0 = blockIdx.x << 6;
  const bool active = (m0 < M);
  const float* arow = Ab + (active ? (long)(m0 + mr) * ldA : 0);

  v8f acc[4] = {};

  for (int kk = 0; kk < K; kk += 32) {
    // ---- stage B panel (32 x 64) into LDS in fragment order ----
    for (int it = tid; it < 16 * 64; it += blockDim.x) {
      const int kp = it >> 6;        // k-pair index 0..15
      const int n  = it & 63;
      const int k  = kp << 1;
      float f0, f1;
      if (BT) {
        const float2 v2 = *(const float2*)(Bb + (long)(n0 + n) * ldB + (kk + k));
        f0 = v2.x; f1 = v2.y;
      } else {
        const float* bp = Bb + (long)(kk + k) * ldB + (n0 + n);
        f0 = bp[0]; f1 = bp[ldB];
      }
      const int ho = ((n >> 4) << 9) + ((k >> 4) << 8) + ((n & 15) << 4) + (k & 15);
      v2h hp = { (_Float16)f0, (_Float16)f1 };
      *(v2h*)&Bs[ho] = hp;           // single ds_store_b32
    }
    __syncthreads();

    if (active) {
      // A fragment (16x32): VGPR v: K = 2v,2v+1 (+16 lanes 16-31)
      v16h af;
#pragma unroll
      for (int v = 0; v < 8; ++v) {
        const int ka = kk + (v << 1) + (g << 4);
        const float2 av = *(const float2*)(arow + ka);
        af[2 * v]     = (_Float16)av.x;
        af[2 * v + 1] = (_Float16)av.y;
      }
#pragma unroll
      for (int s = 0; s < 4; ++s) {
        const v16h bf = *(const v16h*)&Bs[(s << 9) + lane * 16];
        acc[s] = __builtin_amdgcn_wmma_f32_16x16x32_f16(
            false, af, false, bf, (short)0, acc[s], false, false);
      }
    }
    __syncthreads();
  }

  if (active) {
    // C/D layout: VGPR r -> M = m0 + r (lanes 0-15) / m0 + r + 8 (lanes 16-31)
#pragma unroll
    for (int s = 0; s < 4; ++s) {
      const int n = n0 + (s << 4) + mr;
      const float bv = bias ? bias[n] : 0.f;
#pragma unroll
      for (int r = 0; r < 8; ++r) {
        const int m = m0 + r + (g << 3);
        Cb[(long)m * N + n] = acc[s][r] + bv;
      }
    }
  }
}

// kb2[b] = dot(k[b,:], b2)  — one wave per batch row
__global__ __launch_bounds__(32)
void kb2_kernel(const float* __restrict__ k, const float* __restrict__ b2,
                float* __restrict__ kb2)
{
  const int b = blockIdx.x, lane = threadIdx.x;
  const float* kr = k + (long)b * HD;
  float s = 0.f;
  for (int i = lane; i < HD; i += 32) s += kr[i] * b2[i];
  s = wave_sum(s);
  if (lane == 0) kb2[b] = s;
}

// att[b,l] = X[b,l,:]·kw2[b,:] + kb2[b]   (streams all of X: 134 MB)
__global__ __launch_bounds__(256)
void att_kernel(const float* __restrict__ X, const float* __restrict__ kw2,
                const float* __restrict__ kb2, float* __restrict__ att)
{
  const int wave = threadIdx.x >> 5;
  const int lane = threadIdx.x & 31;
  const int row  = blockIdx.x * 8 + wave;   // [0, BS*LCT)
  const int b    = row >> 9;                // row / LCT
  const float4* x4 = (const float4*)(X   + (long)row * HD);
  const float4* k4 = (const float4*)(kw2 + (long)b   * HD);
  float s = 0.f;
#pragma unroll
  for (int i = 0; i < 8; ++i) {
    float4 xv = x4[lane + i * 32];
    float4 kv = k4[lane + i * 32];
    s += xv.x * kv.x + xv.y * kv.y + xv.z * kv.z + xv.w * kv.w;
  }
  s = wave_sum(s);
  if (lane == 0) att[row] = s + kb2[b];
}

// masked softmax over L=512 per (b, phrase); one 256-thread block each
__global__ __launch_bounds__(256)
void softmax_kernel(const float* __restrict__ att,
                    const unsigned char* __restrict__ mask,
                    float* __restrict__ p)
{
  __shared__ float sdata[256];
  const int bp = blockIdx.x;       // b*NPH + ph
  const int b  = bp >> 4;
  const int t  = threadIdx.x;
  const float NEG_INF = -__builtin_inff();
  const float* ar = att + (long)b * LCT;
  const unsigned char* mr = mask + (long)bp * LCT;
  float a0 = mr[t]       ? NEG_INF : ar[t];
  float a1 = mr[t + 256] ? NEG_INF : ar[t + 256];
  sdata[t] = fmaxf(a0, a1); __syncthreads();
  for (int s = 128; s > 0; s >>= 1) {
    if (t < s) sdata[t] = fmaxf(sdata[t], sdata[t + s]);
    __syncthreads();
  }
  const float m = sdata[0]; __syncthreads();
  float e0 = __expf(a0 - m), e1 = __expf(a1 - m);
  sdata[t] = e0 + e1; __syncthreads();
  for (int s = 128; s > 0; s >>= 1) {
    if (t < s) sdata[t] += sdata[t + s];
    __syncthreads();
  }
  const float inv = 1.f / sdata[0];
  float* pr = p + (long)bp * LCT;
  pr[t]       = e0 * inv;
  pr[t + 256] = e1 * inv;
}

// In-place LayerNorm over H=1024 (+ optional residual row X[row/resDiv], ReLU)
__global__ __launch_bounds__(256)
void ln_kernel(float* __restrict__ y, const float* __restrict__ gamma,
               const float* __restrict__ beta, const float* __restrict__ res,
               long resStride, int resDiv, int relu)
{
  __shared__ float sdata[256];
  const long row = blockIdx.x;
  const int  t   = threadIdx.x;
  float4* yr = (float4*)(y + row * HD);
  float4 v = yr[t];
  sdata[t] = v.x + v.y + v.z + v.w; __syncthreads();
  for (int i = 128; i > 0; i >>= 1) { if (t < i) sdata[t] += sdata[t + i]; __syncthreads(); }
  const float mean = sdata[0] * (1.f / HD); __syncthreads();
  const float dx = v.x - mean, dy = v.y - mean, dz = v.z - mean, dw = v.w - mean;
  sdata[t] = dx * dx + dy * dy + dz * dz + dw * dw; __syncthreads();
  for (int i = 128; i > 0; i >>= 1) { if (t < i) sdata[t] += sdata[t + i]; __syncthreads(); }
  const float rstd = rsqrtf(sdata[0] * (1.f / HD) + EPS);
  const float4 gv = ((const float4*)gamma)[t];
  const float4 bv = ((const float4*)beta)[t];
  float4 o = { dx * rstd * gv.x + bv.x, dy * rstd * gv.y + bv.y,
               dz * rstd * gv.z + bv.z, dw * rstd * gv.w + bv.w };
  if (res) {
    const float4 rv = ((const float4*)(res + (row / resDiv) * resStride))[t];
    o.x += rv.x; o.y += rv.y; o.z += rv.z; o.w += rv.w;
  }
  if (relu) {
    o.x = fmaxf(o.x, 0.f); o.y = fmaxf(o.y, 0.f);
    o.z = fmaxf(o.z, 0.f); o.w = fmaxf(o.w, 0.f);
  }
  yr[t] = o;
}

// xcat[row, 0:H]=ctx[row], xcat[row, H:2H]=lq[row]
__global__ __launch_bounds__(256)
void concat_kernel(const float* __restrict__ ctx, const float* __restrict__ lq,
                   float* __restrict__ xcat)
{
  const long idx4 = (long)blockIdx.x * 256 + threadIdx.x;  // float4 index over 1024x2048
  const long row  = idx4 >> 9;                             // 512 float4 per row
  const int  c4   = (int)(idx4 & 511);
  const float4 src = (c4 < 256)
      ? ((const float4*)(ctx + row * HD))[c4]
      : ((const float4*)(lq  + row * HD))[c4 - 256];
  ((float4*)(xcat + row * 2 * HD))[c4] = src;
}

// out_a[r,b,h] = x[b,p,h] + qe[n,h];  out_b[r,b,h] = x[b,p,h] + qe[n,H+h]
// where r = p*NQ + n.  134 MB of stores — pure bandwidth.
__global__ __launch_bounds__(256)
void expand_out(const float* __restrict__ x, const float* __restrict__ qe,
                float* __restrict__ outA, float* __restrict__ outB)
{
  const int b = blockIdx.x & 63;
  const int r = blockIdx.x >> 6;     // 0..255
  const int p = r >> 4;
  const int n = r & 15;
  const int t = threadIdx.x;         // 256 threads * float4 = 1024 floats
  const float4* xr = (const float4*)(x + ((long)b * NPH + p) * HD);
  const float4* qa = (const float4*)(qe + (long)n * 2 * HD);
  const float4* qb = qa + HD / 4;
  const float4 xv = xr[t], qav = qa[t], qbv = qb[t];
  const long orow = ((long)r * BS + b) * HD;
  float4 oa = { xv.x + qav.x, xv.y + qav.y, xv.z + qav.z, xv.w + qav.w };
  float4 ob = { xv.x + qbv.x, xv.y + qbv.y, xv.z + qbv.z, xv.w + qbv.w };
  ((float4*)(outA + orow))[t] = oa;
  ((float4*)(outB + orow))[t] = ob;
}

extern "C" void kernel_launch(void* const* d_in, const int* in_sizes, int n_in,
                              void* d_out, int out_size, void* d_ws, size_t ws_size,
                              hipStream_t stream)
{
  (void)in_sizes; (void)n_in; (void)out_size; (void)ws_size;
  const float* X    = (const float*)d_in[0];   // [BS,LCT,HD]
  const float* LQ   = (const float*)d_in[1];   // [BS,NPH,HD]
  const unsigned char* MASK = (const unsigned char*)d_in[2];   // bool [BS,NPH,LCT]
  const float* QE   = (const float*)d_in[3];   // [NQ, 2H]
  const float* W1   = (const float*)d_in[4];   const float* b1  = (const float*)d_in[5];
  const float* W2   = (const float*)d_in[6];   const float* b2  = (const float*)d_in[7];
  const float* W3   = (const float*)d_in[8];   const float* b3  = (const float*)d_in[9];
  const float* Wc   = (const float*)d_in[10];  const float* bc  = (const float*)d_in[11];
  const float* gc   = (const float*)d_in[12];  const float* bec = (const float*)d_in[13];
  const float* Wf1  = (const float*)d_in[14];  const float* bf1 = (const float*)d_in[15];
  const float* gf1  = (const float*)d_in[16];  const float* bef1= (const float*)d_in[17];
  const float* Wf2  = (const float*)d_in[18];  const float* bf2 = (const float*)d_in[19];
  const float* gf2  = (const float*)d_in[20];  const float* bef2= (const float*)d_in[21];

  // Workspace layout (floats), ~32 MB total
  float* ws   = (float*)d_ws;
  float* kbuf = ws;                          // [BS,HD]
  float* kw2b = kbuf + BS * HD;              // [BS,HD]
  float* kb2b = kw2b + BS * HD;              // [BS]
  float* attb = kb2b + BS;                   // [BS,LCT]
  float* pbuf = attb + BS * LCT;             // [BS,NPH,LCT]
  float* ubuf = pbuf + BS * NPH * LCT;       // [BS,NPH,HD]
  float* c0   = ubuf + (long)BS * NPH * HD;  // [BS*NPH,HD]
  float* c1   = c0   + (long)BS * NPH * HD;  // [BS*NPH,HD]
  float* xcat = c1   + (long)BS * NPH * HD;  // [BS*NPH,2H]
  float* h1   = xcat + (long)BS * NPH * 2 * HD;
  float* h2   = h1   + (long)BS * NPH * HD;

  // 1) k = X[:,0,:] @ W1 + b1          (A rows are token-0 rows: ldA = LCT*HD)
  gemm_wmma_f16<false><<<dim3(HD / 64, 1, 1), 128, 0, stream>>>(
      X, W1, b1, kbuf, BS, HD, HD, (long)LCT * HD, HD, 0, 0, 0);
  // 2) kw2 = k @ W2ᵀ  (fold W2 into k; kills the 34-GMAC q-GEMM)
  gemm_wmma_f16<true><<<dim3(HD / 64, 1, 1), 128, 0, stream>>>(
      kbuf, W2, nullptr, kw2b, BS, HD, HD, HD, HD, 0, 0, 0);
  // 3) kb2[b] = k[b]·b2
  kb2_kernel<<<BS, 32, 0, stream>>>(kbuf, b2, kb2b);
  // 4) att[b,l] = X[b,l]·kw2[b] + kb2[b]   (streams X once)
  att_kernel<<<BS * LCT / 8, 256, 0, stream>>>(X, kw2b, kb2b, attb);
  // 5) masked softmax -> p
  softmax_kernel<<<BS * NPH, 256, 0, stream>>>(attb, MASK, pbuf);
  // 6) u = p @ X (batched; X second pass is L2-resident; kills the v-GEMM)
  gemm_wmma_f16<false><<<dim3(HD / 64, 1, BS), 32, 0, stream>>>(
      pbuf, X, nullptr, ubuf, NPH, HD, LCT, LCT, HD,
      (long)NPH * LCT, (long)LCT * HD, (long)NPH * HD);
  // 7) ctx0 = u @ W3 + b3
  gemm_wmma_f16<false><<<dim3(HD / 64, BS * NPH / 64, 1), 128, 0, stream>>>(
      ubuf, W3, b3, c0, BS * NPH, HD, HD, HD, HD, 0, 0, 0);
  // 8) c1 = ctx0 @ Wc + bc
  gemm_wmma_f16<false><<<dim3(HD / 64, BS * NPH / 64, 1), 128, 0, stream>>>(
      c0, Wc, bc, c1, BS * NPH, HD, HD, HD, HD, 0, 0, 0);
  // 9) c1 = LN(c1)*gc+betac + X[b,0,:]
  ln_kernel<<<BS * NPH, 256, 0, stream>>>(c1, gc, bec, X, (long)LCT * HD, NPH, 0);
  // 10) xcat = [c1, LQ]
  concat_kernel<<<2048, 256, 0, stream>>>(c1, LQ, xcat);
  // 11) h1 = xcat @ Wf1 + bf1  (K = 2048)
  gemm_wmma_f16<false><<<dim3(HD / 64, BS * NPH / 64, 1), 128, 0, stream>>>(
      xcat, Wf1, bf1, h1, BS * NPH, HD, 2 * HD, 2 * HD, HD, 0, 0, 0);
  // 12) h1 = relu(LN(h1))
  ln_kernel<<<BS * NPH, 256, 0, stream>>>(h1, gf1, bef1, nullptr, 0, 1, 1);
  // 13) h2 = h1 @ Wf2 + bf2
  gemm_wmma_f16<false><<<dim3(HD / 64, BS * NPH / 64, 1), 128, 0, stream>>>(
      h1, Wf2, bf2, h2, BS * NPH, HD, HD, HD, HD, 0, 0, 0);
  // 14) h2 = relu(LN(h2))
  ln_kernel<<<BS * NPH, 256, 0, stream>>>(h2, gf2, bef2, nullptr, 0, 1, 1);
  // 15) expand to [NPH*NQ, BS, H] x 2 with query_embed add
  float* outA = (float*)d_out;
  float* outB = outA + (long)NPH * NQ * BS * HD;
  expand_out<<<NPH * NQ * BS, 256, 0, stream>>>(h2, QE, outA, outB);
}
